// Seastar_EglRelGraphConv_6399501271225
// MI455X (gfx1250) — compile-verified, matrix operand order checked by reading, not run on hardware
//
#include <hip/hip_runtime.h>

// ---------------------------------------------------------------------------
// R-GCN (basis decomposition) fused kernel for MI455X (gfx1250, wave32, WMMA)
//
//   W[r]      = sum_b w_comp[r,b] * weight[b]            (tiny)
//   msg[e]    = (x[src[e]] @ W[etype[e]]) * norm[e]      (13.1 GFLOP, bf16 WMMA)
//   out[n]    = h_bias + sum_{e: dst[e]==n} msg[e]       (f32 atomics, L2 resident)
//
// x is pre-converted to bf16 once (each node is a src ~16x on average), edges
// are counting-sorted by relation, and each wave32 runs a 16-edge x 64-out
// tile with a single W[r]: 8x v_wmma_f32_16x16x32_bf16 per tile, with A and B
// operands loaded directly as b128s (no conversion VALU in the hot path).
// ---------------------------------------------------------------------------

#define N_NODES   100000
#define N_EDGES   1600000
#define FEAT      64
#define NUM_RELS  32
#define NUM_BASES 8
#define MAX_TILES (N_EDGES / 16 + NUM_RELS)   // 100032

typedef __attribute__((ext_vector_type(16))) __bf16         v16bf;
typedef __attribute__((ext_vector_type(16))) unsigned short v16us;
typedef __attribute__((ext_vector_type(8)))  float          v8f;

union ABf16 { v16us u; v16bf b; uint4 q[2]; };

__device__ inline unsigned short f32_to_bf16(float f) {
  unsigned u = __builtin_bit_cast(unsigned, f);
  unsigned r = u + 0x7FFFu + ((u >> 16) & 1u);   // round-to-nearest-even
  return (unsigned short)(r >> 16);
}

// ---- out[n][o] = h_bias[o] (bias + zero-init in one pass) ------------------
__global__ void rgcn_init_out(float* __restrict__ out, const float* __restrict__ h_bias) {
  int i = blockIdx.x * blockDim.x + threadIdx.x;
  if (i < N_NODES * FEAT) out[i] = h_bias[i & (FEAT - 1)];
}

// ---- zero the counter/offset region ---------------------------------------
__global__ void rgcn_zero_ints(int* __restrict__ ints) {
  ints[threadIdx.x] = 0;   // 256 ints
}

// ---- x (f32) -> x_bf (bf16, row-major), 8 elements per thread --------------
__global__ void rgcn_xcast(const float* __restrict__ x, unsigned short* __restrict__ xbf) {
  int i = (blockIdx.x * blockDim.x + threadIdx.x) * 8;
  if (i < N_NODES * FEAT) {
    float4 a = *(const float4*)(x + i);
    float4 b = *(const float4*)(x + i + 4);
    ushort4 lo, hi;
    lo.x = f32_to_bf16(a.x); lo.y = f32_to_bf16(a.y);
    lo.z = f32_to_bf16(a.z); lo.w = f32_to_bf16(a.w);
    hi.x = f32_to_bf16(b.x); hi.y = f32_to_bf16(b.y);
    hi.z = f32_to_bf16(b.z); hi.w = f32_to_bf16(b.w);
    *(ushort4*)(xbf + i)     = lo;
    *(ushort4*)(xbf + i + 4) = hi;
  }
}

// ---- basis mix, written in WMMA-B lane-swizzled bf16 layout ----------------
// Wswz index: (((r*2 + kt)*4 + nt)*32 + lane)*16 + h
//   N = nt*16 + (lane & 15)
//   K = kt*32 + ((lane>>4)&1)*8 + (h<8 ? h : h+8)
__global__ void rgcn_basis(const float* __restrict__ weight,
                           const float* __restrict__ w_comp,
                           unsigned short* __restrict__ Wswz) {
  int r = blockIdx.x;            // 32 blocks
  int t = threadIdx.x;           // 256 threads
  int kt   = t >> 7;
  int nt   = (t >> 5) & 3;
  int lane = t & 31;

  float wc[NUM_BASES];
#pragma unroll
  for (int b = 0; b < NUM_BASES; ++b) wc[b] = w_comp[r * NUM_BASES + b];

  unsigned short* dstp = Wswz + ((((r * 2 + kt) * 4 + nt) * 32 + lane) * 16);
  int kbase = kt * 32 + ((lane >> 4) & 1) * 8;
  int n     = nt * 16 + (lane & 15);
#pragma unroll
  for (int h = 0; h < 16; ++h) {
    int K = kbase + (h < 8 ? h : h + 8);
    float s = 0.f;
#pragma unroll
    for (int b = 0; b < NUM_BASES; ++b)
      s += wc[b] * weight[b * (FEAT * FEAT) + K * FEAT + n];
    dstp[h] = f32_to_bf16(s);
  }
}

// ---- relation histogram ----------------------------------------------------
__global__ void rgcn_hist(const int* __restrict__ etypes, int* __restrict__ hist) {
  int e = blockIdx.x * blockDim.x + threadIdx.x;
  if (e < N_EDGES) atomicAdd(&hist[etypes[e]], 1);
}

// ---- 32-wide scan: edge offsets + tile offsets -----------------------------
__global__ void rgcn_scan(const int* __restrict__ hist,
                          int* __restrict__ offsets, int* __restrict__ tileoff) {
  int acc = 0, tacc = 0;
  offsets[0] = 0; tileoff[0] = 0;
  for (int r = 0; r < NUM_RELS; ++r) {
    int cnt = hist[r];
    acc  += cnt;             offsets[r + 1] = acc;
    tacc += (cnt + 15) >> 4; tileoff[r + 1] = tacc;
  }
}

// ---- scatter edges into relation-sorted permutation ------------------------
__global__ void rgcn_scatter(const int* __restrict__ etypes,
                             const int* __restrict__ offsets,
                             int* __restrict__ counters, int* __restrict__ perm) {
  int e = blockIdx.x * blockDim.x + threadIdx.x;
  if (e < N_EDGES) {
    int r = etypes[e];
    int pos = offsets[r] + atomicAdd(&counters[r], 1);
    perm[pos] = e;
  }
}

// ---- main WMMA kernel: one wave32 per 16-edge x 64-out tile ----------------
__global__ __launch_bounds__(32)
void rgcn_wmma(const unsigned short* __restrict__ xbf,
               const float* __restrict__ norm,
               const int* __restrict__ src,
               const int* __restrict__ dst,
               const unsigned short* __restrict__ Wswz,
               const int* __restrict__ offsets,
               const int* __restrict__ tileoff,
               const int* __restrict__ perm,
               float* __restrict__ out) {
  int tile  = blockIdx.x;
  int total = tileoff[NUM_RELS];
  if (tile >= total) return;                 // wave-uniform

  // find relation for this tile (<=32-step scalar walk)
  int r = 0;
#pragma unroll 1
  while (tileoff[r + 1] <= tile) ++r;

  int ebase = offsets[r] + ((tile - tileoff[r]) << 4);
  int ecnt  = offsets[r + 1] - ebase;
  if (ecnt > 16) ecnt = 16;
  int elast = ecnt - 1;

  int lane = threadIdx.x;
  int m    = lane & 15;                      // A-matrix row this lane feeds
  // clamped gather: always a valid load, tail rows hold duplicate finite data
  int mc   = m < ecnt ? m : elast;
  int ea   = perm[ebase + mc];
  const unsigned short* xrow = xbf + (size_t)src[ea] * FEAT;
  int csel = (lane >> 4) & 1;                // column-half select (ISA A layout)

  v8f acc[4] = {};                           // 4 N-tiles of f32 16x16 C/D

#pragma unroll
  for (int kt = 0; kt < 2; ++kt) {
    ABf16 A;
    int cb = kt * 32 + csel * 8;             // multiple of 8 -> 16B-aligned
    A.q[0] = *(const uint4*)(xrow + cb);          // K = cb   .. cb+7
    A.q[1] = *(const uint4*)(xrow + cb + 16);     // K = cb+16.. cb+23
#pragma unroll
    for (int nt = 0; nt < 4; ++nt) {
      ABf16 B;
      const uint4* bp = (const uint4*)(Wswz + ((((r * 2 + kt) * 4 + nt) * 32 + lane) * 16));
      B.q[0] = bp[0];
      B.q[1] = bp[1];
      acc[nt] = __builtin_amdgcn_wmma_f32_16x16x32_bf16(
          false, A.b, false, B.b, (short)0, acc[nt], false, false);
    }
  }

  // branchless epilogue: per D-row, one base pointer + 4 atomics at imm offsets
  int nlocal = lane & 15;
  int mhi    = (lane >> 4) * 8;
#pragma unroll
  for (int j = 0; j < 8; ++j) {
    int M  = mhi + j;
    int ec = M < ecnt ? M : elast;           // clamped -> always-valid loads
    int e  = perm[ebase + ec];
    float s = (M < ecnt) ? norm[e] : 0.f;    // tail rows add 0.0f (no-op)
    float* p = out + (size_t)dst[e] * FEAT + nlocal;
    atomicAdd(p,      acc[0][j] * s);
    atomicAdd(p + 16, acc[1][j] * s);
    atomicAdd(p + 32, acc[2][j] * s);
    atomicAdd(p + 48, acc[3][j] * s);
  }
}

// ---------------------------------------------------------------------------
extern "C" void kernel_launch(void* const* d_in, const int* in_sizes, int n_in,
                              void* d_out, int out_size, void* d_ws, size_t ws_size,
                              hipStream_t stream) {
  const float* x      = (const float*)d_in[0];
  const float* weight = (const float*)d_in[1];
  const float* w_comp = (const float*)d_in[2];
  const float* h_bias = (const float*)d_in[3];
  const float* norm   = (const float*)d_in[4];
  const int*   src    = (const int*)d_in[5];
  const int*   dst    = (const int*)d_in[6];
  const int*   etypes = (const int*)d_in[7];
  float*       out    = (float*)d_out;

  // workspace: [256KB swizzled W][1KB counters][6.4MB perm][12.8MB x_bf]
  unsigned short* Wswz = (unsigned short*)d_ws;
  int* ints     = (int*)((char*)d_ws + 262144);
  int* hist     = ints;          // 32
  int* offsets  = ints + 32;     // 33
  int* counters = ints + 72;     // 32
  int* tileoff  = ints + 112;    // 33
  int* perm     = (int*)((char*)d_ws + 263168);
  unsigned short* xbf = (unsigned short*)((char*)d_ws + 6664192);

  rgcn_init_out<<<(N_NODES * FEAT + 255) / 256, 256, 0, stream>>>(out, h_bias);
  rgcn_zero_ints<<<1, 256, 0, stream>>>(ints);
  rgcn_xcast<<<(N_NODES * FEAT / 8 + 255) / 256, 256, 0, stream>>>(x, xbf);
  rgcn_basis<<<NUM_RELS, 256, 0, stream>>>(weight, w_comp, Wswz);
  rgcn_hist<<<(N_EDGES + 255) / 256, 256, 0, stream>>>(etypes, hist);
  rgcn_scan<<<1, 1, 0, stream>>>(hist, offsets, tileoff);
  rgcn_scatter<<<(N_EDGES + 255) / 256, 256, 0, stream>>>(etypes, offsets, counters, perm);
  rgcn_wmma<<<MAX_TILES, 32, 0, stream>>>(xbf, norm, src, dst, Wswz, offsets, tileoff, perm, out);
}
